// CoxLoss_65515431133625
// MI455X (gfx1250) — matrix operand order repaired and checked by reading, not compiled, not run on hardware
//
#include <hip/hip_runtime.h>

// CDNA5 / gfx1250 — wave32, WMMA 16x16x32 f16 with f32 accumulate.
typedef __attribute__((ext_vector_type(16))) _Float16 v16h;
typedef __attribute__((ext_vector_type(8)))  float    v8f;

#define NN       8192
#define THREADS  256           // 8 wave32s per block
#define JP       4             // j-dimension partitions (partial risk sums)
#define JBP      2048          // j elements per partition/block (16KB LDS stage)
#define ROWSPB   256           // 8 waves * 2 row tiles * 16 rows
// grid = (NN/ROWSPB, JP) = (32, 4) = 128 blocks

// Partial risk sums: riskp[p*NN + i] = sum_{j in slice p} (surv[j]>=surv[i])*exp(theta[j])
// Mask tile (exact 0/1 in f16) -> A ; exp(theta) split hi/lo f16 -> B ;
// two v_wmma_f32_16x16x32_f16 per 16x32 tile into one f32 accumulator.
// Each wave owns TWO 16-row tiles: the staged j-data (A-source survtime and B)
// is row-tile independent, so 8 ds_load_b128 feed 4 WMMAs instead of 2.
__global__ __launch_bounds__(THREADS) void cox_risk_wmma(
    const float* __restrict__ theta,
    const float* __restrict__ surv,
    float* __restrict__ riskp)
{
  __shared__ __align__(32) float    s_surv[JBP];
  __shared__ __align__(32) _Float16 s_ehi[JBP];
  __shared__ __align__(32) _Float16 s_elo[JBP];

  const int tid  = threadIdx.x;
  const int lane = tid & 31;
  const int wave = tid >> 5;
  const int ibase0 = blockIdx.x * ROWSPB + wave * 32;   // first 16-row tile
  const int ibase1 = ibase0 + 16;                       // second 16-row tile
  const int j0     = blockIdx.y * JBP;                  // this block's j slice

  // Per-lane row values (lanes 0-15 -> rows 0-15, lanes 16-31 same rows).
  const float si0 = surv[ibase0 + (lane & 15)];
  const float si1 = surv[ibase1 + (lane & 15)];
  const int koff  = (lane >> 4) * 16;   // A-matrix K sub-offset for this lane half
  const int kb    = (lane >> 4) * 16;   // B-matrix K sub-offset for this lane half

  // Stage survtime + double-f16 split of exp(theta) into LDS (coalesced, uniform).
  for (int t = tid; t < JBP; t += THREADS) {
    const int j = j0 + t;
    const float e  = expf(theta[j]);
    const _Float16 ehi = (_Float16)e;
    const _Float16 elo = (_Float16)(e - (float)ehi);
    s_surv[t] = surv[j];
    s_ehi[t]  = ehi;
    s_elo[t]  = elo;
  }
  __syncthreads();

  v8f c0 = {};   // f32 accumulator, row tile 0 (gets e_hi and e_lo contributions)
  v8f c1 = {};   // f32 accumulator, row tile 1

  for (int jc = 0; jc < JBP; jc += 32) {
    // A-source survtime, shared by both row tiles. Contiguous-K lane layout used here (not ISA table order):
    //  elem t   -> K = koff + t        (t = 0..7)
    //  elem 8+t -> K = koff + 8 + t
    const float* sp0 = &s_surv[jc + koff];
    const float* sp1 = &s_surv[jc + koff + 8];
    float sj0[8], sj1[8];
#pragma unroll
    for (int t = 0; t < 8; ++t) { sj0[t] = sp0[t]; sj1[t] = sp1[t]; }

    // Build the two 16x32 f16 mask tiles (exact 0/1).
    v16h a0, a1;
#pragma unroll
    for (int t = 0; t < 8; ++t) {
      a0[t]     = (sj0[t] >= si0) ? (_Float16)1.0f : (_Float16)0.0f;
      a0[8 + t] = (sj1[t] >= si0) ? (_Float16)1.0f : (_Float16)0.0f;
      a1[t]     = (sj0[t] >= si1) ? (_Float16)1.0f : (_Float16)0.0f;
      a1[8 + t] = (sj1[t] >= si1) ? (_Float16)1.0f : (_Float16)0.0f;
    }

    // B (32x16 f16): every column is e[j0+jc+k]; lane needs halves
    // K = kb .. kb+15 -> one aligned 32B LDS vector load each. Reused by both tiles.
    const v16h bh = *(const v16h*)&s_ehi[jc + kb];
    const v16h bl = *(const v16h*)&s_elo[jc + kb];

    c0 = __builtin_amdgcn_wmma_f32_16x16x32_f16(false, a0, false, bh,
                                                (short)0, c0, false, false);
    c0 = __builtin_amdgcn_wmma_f32_16x16x32_f16(false, a0, false, bl,
                                                (short)0, c0, false, false);
    c1 = __builtin_amdgcn_wmma_f32_16x16x32_f16(false, a1, false, bh,
                                                (short)0, c1, false, false);
    c1 = __builtin_amdgcn_wmma_f32_16x16x32_f16(false, a1, false, bl,
                                                (short)0, c1, false, false);
  }

  // D layout: lanes 0-15 hold rows M=r (VGPR r), lanes 16-31 hold M=8+r.
  // All 16 columns of D are identical (B columns replicated) -> one lane per
  // half-wave writes its 8 row sums into this partition's partial-sum slice.
  float* rp = riskp + blockIdx.y * NN;
  if ((lane & 15) == 0) {
    const int mbase = (lane >> 4) * 8;
#pragma unroll
    for (int r = 0; r < 8; ++r) {
      rp[ibase0 + mbase + r] = c0[r];
      rp[ibase1 + mbase + r] = c1[r];
    }
  }
}

// loss = -(1/N) * sum_i (theta_i - log(sum_p riskp[p][i])) * censor_i
// Single-block tree reduction; fixed summation order -> deterministic.
__global__ __launch_bounds__(256) void cox_finalize(
    const float* __restrict__ theta,
    const float* __restrict__ censor,
    const float* __restrict__ riskp,
    float* __restrict__ out)
{
  __shared__ float red[256];
  const int tid = threadIdx.x;
  float acc = 0.0f;
  for (int i = tid; i < NN; i += 256) {
    float r = riskp[i] + riskp[NN + i] + riskp[2 * NN + i] + riskp[3 * NN + i];
    acc += (theta[i] - logf(r)) * censor[i];
  }
  red[tid] = acc;
  __syncthreads();
  for (int s = 128; s > 0; s >>= 1) {
    if (tid < s) red[tid] += red[tid + s];
    __syncthreads();
  }
  if (tid == 0) out[0] = -red[0] / (float)NN;
}

extern "C" void kernel_launch(void* const* d_in, const int* in_sizes, int n_in,
                              void* d_out, int out_size, void* d_ws, size_t ws_size,
                              hipStream_t stream) {
  const float* theta  = (const float*)d_in[0];  // hazard_pred (N,1) flat
  const float* surv   = (const float*)d_in[1];  // survtime (N)
  const float* censor = (const float*)d_in[2];  // censor (N)
  float* riskp = (float*)d_ws;                  // JP * N floats of scratch (128 KB)

  dim3 grid(NN / ROWSPB, JP);
  cox_risk_wmma<<<grid, THREADS, 0, stream>>>(theta, surv, riskp);
  cox_finalize<<<1, 256, 0, stream>>>(theta, censor, riskp, (float*)d_out);
}
